// ResM_46205258170734
// MI455X (gfx1250) — compile-verified, matrix operand order checked by reading, not run on hardware
//
#include <hip/hip_runtime.h>
#include <hip/hip_bf16.h>
#include <math.h>

// ---------------- problem constants ----------------
#define T_TOK  16384      // B*S
#define IN_DIMC 3072
#define V_DIMC  768
#define HEADSC  4
#define K_DIMC  256
#define HALFC   128
#define NKEYSC  128
#define KNNC    32
#define QDIMC   1024      // HEADS*K_DIM
#define NCATC   1792      // V_DIM + QDIM

typedef __attribute__((ext_vector_type(16))) __bf16 v16bf;
typedef __attribute__((ext_vector_type(8)))  float  v8f;

// CDNA5 async global->LDS path (ASYNCcnt-tracked), with safe fallback
#if defined(__gfx1250__) && __has_builtin(__builtin_amdgcn_global_load_async_to_lds_b128) && __has_builtin(__builtin_amdgcn_s_wait_asynccnt)
#define USE_ASYNC_LDS 1
#else
#define USE_ASYNC_LDS 0
#endif

#if USE_ASYNC_LDS
// builtin signature: (v4i addrspace(1)*, v4i addrspace(3)*, imm int offset, imm int cpol)
typedef __attribute__((__vector_size__(4 * sizeof(int)))) int async_v4i;
typedef __attribute__((address_space(1))) async_v4i* async_gptr;
typedef __attribute__((address_space(3))) async_v4i* async_lptr;
#endif

__device__ __forceinline__ void copy16_to_lds(const __bf16* g, __bf16* l) {
#if USE_ASYNC_LDS
  __builtin_amdgcn_global_load_async_to_lds_b128((async_gptr)(g), (async_lptr)(l), 0, 0);
#else
  *reinterpret_cast<uint4*>(l) = *reinterpret_cast<const uint4*>(g);
#endif
}

__device__ __forceinline__ void wait_tiles_landed() {
#if USE_ASYNC_LDS
  __builtin_amdgcn_s_wait_asynccnt(0);
#endif
  __syncthreads();
}

// ---------------- conversion kernels ----------------
__global__ void f32_to_bf16_kernel(const float* __restrict__ in,
                                   __bf16* __restrict__ out, int n) {
  int i = blockIdx.x * 256 + threadIdx.x;
  if (i < n) out[i] = (__bf16)in[i];
}

// Wt[n][k] = n<768 ? W_dense[k][n] : W_q[k][n-768]   (bf16, K-contiguous rows)
__global__ void build_wt_kernel(const float* __restrict__ Wd,
                                const float* __restrict__ Wq,
                                __bf16* __restrict__ Wt) {
  int i = blockIdx.x * 256 + threadIdx.x;
  if (i >= NCATC * IN_DIMC) return;
  int n = i / IN_DIMC, k = i % IN_DIMC;
  float v = (n < V_DIMC) ? Wd[(size_t)k * V_DIMC + n]
                         : Wq[(size_t)k * QDIMC + (n - V_DIMC)];
  Wt[i] = (__bf16)v;
}

__global__ void build_bias_kernel(const float* __restrict__ bd,
                                  const float* __restrict__ bq,
                                  float* __restrict__ bias) {
  int i = blockIdx.x * 256 + threadIdx.x;
  if (i < NCATC) bias[i] = (i < V_DIMC) ? bd[i] : bq[i - V_DIMC];
}

// ---------------- bf16 WMMA GEMM core (double-buffered LDS tiles) ----------------
#define BM 128
#define BN 128
#define BKK 32
#define LDS_STRIDE 40   // bf16 elems; 80B rows keep 16B alignment, dodge bank conflicts

template <class EpiFn>
__device__ __forceinline__ void gemm_core_bf16(const __bf16* __restrict__ A, int lda,
                                               const __bf16* __restrict__ Bt, int ldb,
                                               int K, int mbase, int nbase, EpiFn epi) {
  __shared__ __align__(16) __bf16 As[2][BM * LDS_STRIDE];
  __shared__ __align__(16) __bf16 Bs[2][BN * LDS_STRIDE];

  const int tid  = threadIdx.x;
  const int lane = tid & 31;
  const int wid  = tid >> 5;   // 0..7
  const int wm   = wid >> 2;   // 0..1 : wave row (64 rows)
  const int wn   = wid & 3;    // 0..3 : wave col (32 cols)
  const int halfSel = lane >> 4;
  const int r16     = lane & 15;
  const int kb0     = halfSel * 8;   // per ISA 16-bit A/B fragment K layout
  const int kb1     = kb0 + 16;
  const int crow = tid >> 2;          // copy coords: 2x (A,B) 16B chunks/thread
  const int cseg = (tid & 3) * 8;

  v8f acc[4][2];
  const v8f vzero = {0.f,0.f,0.f,0.f,0.f,0.f,0.f,0.f};
#pragma unroll
  for (int i = 0; i < 4; ++i)
#pragma unroll
    for (int j = 0; j < 2; ++j) acc[i][j] = vzero;

  auto loadTile = [&](int kc, int buf) {
    copy16_to_lds(A  + (size_t)(mbase + crow)      * lda + kc + cseg,
                  &As[buf][ crow       * LDS_STRIDE + cseg]);
    copy16_to_lds(A  + (size_t)(mbase + crow + 64) * lda + kc + cseg,
                  &As[buf][(crow + 64) * LDS_STRIDE + cseg]);
    copy16_to_lds(Bt + (size_t)(nbase + crow)      * ldb + kc + cseg,
                  &Bs[buf][ crow       * LDS_STRIDE + cseg]);
    copy16_to_lds(Bt + (size_t)(nbase + crow + 64) * ldb + kc + cseg,
                  &Bs[buf][(crow + 64) * LDS_STRIDE + cseg]);
  };

  const int nIter = K / BKK;
  loadTile(0, 0);
  for (int it = 0; it < nIter; ++it) {
    const int buf = it & 1;
    wait_tiles_landed();                       // s_wait_asynccnt 0 + barrier
    if (it + 1 < nIter) loadTile((it + 1) * BKK, buf ^ 1);

    v16bf afrag[4], bfrag[2];
#pragma unroll
    for (int i = 0; i < 4; ++i) {
      int row = wm * 64 + i * 16 + r16;
      union { uint4 u[2]; v16bf v; } t;
      t.u[0] = *reinterpret_cast<const uint4*>(&As[buf][row * LDS_STRIDE + kb0]);
      t.u[1] = *reinterpret_cast<const uint4*>(&As[buf][row * LDS_STRIDE + kb1]);
      afrag[i] = t.v;
    }
#pragma unroll
    for (int j = 0; j < 2; ++j) {
      int row = wn * 32 + j * 16 + r16;
      union { uint4 u[2]; v16bf v; } t;
      t.u[0] = *reinterpret_cast<const uint4*>(&Bs[buf][row * LDS_STRIDE + kb0]);
      t.u[1] = *reinterpret_cast<const uint4*>(&Bs[buf][row * LDS_STRIDE + kb1]);
      bfrag[j] = t.v;
    }
#pragma unroll
    for (int i = 0; i < 4; ++i)
#pragma unroll
      for (int j = 0; j < 2; ++j)
        acc[i][j] = __builtin_amdgcn_wmma_f32_16x16x32_bf16(
            false, afrag[i], false, bfrag[j], (short)0, acc[i][j], false, false);
  }

  // C/D layout: VGPR r: lanes 0-15 -> M=r, lanes 16-31 -> M=8+r ; N = lane&15
#pragma unroll
  for (int i = 0; i < 4; ++i)
#pragma unroll
    for (int j = 0; j < 2; ++j) {
      int n = nbase + wn * 32 + j * 16 + r16;
#pragma unroll
      for (int r = 0; r < 8; ++r) {
        int m = mbase + wm * 64 + i * 16 + r + halfSel * 8;
        epi(m, n, acc[i][j][r]);
      }
    }
}

// main GEMM: (T x 3072) @ (3072 x 1792) + bias; dense cols -> fp32, q cols -> bf16
__launch_bounds__(256)
__global__ void gemm_main_kernel(const __bf16* __restrict__ XB,
                                 const __bf16* __restrict__ WT,
                                 const float* __restrict__ bias,
                                 float* __restrict__ DENSE,
                                 __bf16* __restrict__ QB) {
  const int mbase = blockIdx.x * BM, nbase = blockIdx.y * BN;
  gemm_core_bf16(XB, IN_DIMC, WT, IN_DIMC, IN_DIMC, mbase, nbase,
    [&](int m, int n, float v) {
      v += bias[n];
      if (n < V_DIMC) DENSE[(size_t)m * V_DIMC + n] = v;
      else            QB[(size_t)m * QDIMC + (n - V_DIMC)] = (__bf16)v;
    });
}

// score GEMMs, batched over z = h*2+sub: s = q[z] (T x 128) @ keys[z]^T (128 x 128)
__launch_bounds__(256)
__global__ void gemm_score_kernel(const __bf16* __restrict__ QB,
                                  const __bf16* __restrict__ KB,
                                  float* __restrict__ SB) {
  const int z = blockIdx.z;
  const int mbase = blockIdx.x * BM;
  float* Cz = SB + (size_t)z * T_TOK * HALFC;
  gemm_core_bf16(QB + z * HALFC, QDIMC,
                 KB + (size_t)z * NKEYSC * HALFC, HALFC,
                 HALFC, mbase, 0,
    [&](int m, int n, float v) { Cz[(size_t)m * HALFC + n] = v; });
}

// ---------------- top-k / cartesian / softmax (one wave32 per (token,head)) ----------------
__device__ inline void top32_of_128(float v0, float v1, float v2, float v3,
                                    int lane, float& outv, int& outi) {
  outv = 0.0f; outi = 0;
  for (int iter = 0; iter < 32; ++iter) {
    float m = v0; int mi = lane;
    if (v1 > m) { m = v1; mi = lane + 32; }
    if (v2 > m) { m = v2; mi = lane + 64; }
    if (v3 > m) { m = v3; mi = lane + 96; }
#pragma unroll
    for (int off = 16; off; off >>= 1) {
      float ov = __shfl_xor(m, off, 32);
      int   oi = __shfl_xor(mi, off, 32);
      if (ov > m || (ov == m && oi < mi)) { m = ov; mi = oi; }
    }
    if (lane == (mi & 31)) {
      int slot = mi >> 5;
      if      (slot == 0) v0 = -__builtin_inff();
      else if (slot == 1) v1 = -__builtin_inff();
      else if (slot == 2) v2 = -__builtin_inff();
      else                v3 = -__builtin_inff();
    }
    if (lane == iter) { outv = m; outi = mi; }
  }
}

__launch_bounds__(256)
__global__ void pkm_topk_kernel(const float* __restrict__ S,  // (H,2,T,128)
                                float* __restrict__ Wgt,      // (T,H,32)
                                int*   __restrict__ Idx) {    // (T,H,32)
  __shared__ float s2v[8][32];
  __shared__ int   s1i[8][32];
  __shared__ int   s2i[8][32];
  const int tid = threadIdx.x, lane = tid & 31, wv = tid >> 5;
  const int g = blockIdx.x * 8 + wv;     // token-head id
  const int t = g >> 2, h = g & 3;

  const float* s1p = S + ((size_t)(h * 2 + 0) * T_TOK + t) * HALFC;
  const float* s2p = S + ((size_t)(h * 2 + 1) * T_TOK + t) * HALFC;

  float v1l; int i1l;
  top32_of_128(s1p[lane], s1p[lane + 32], s1p[lane + 64], s1p[lane + 96], lane, v1l, i1l);
  float v2l; int i2l;
  top32_of_128(s2p[lane], s2p[lane + 32], s2p[lane + 64], s2p[lane + 96], lane, v2l, i2l);

  s1i[wv][lane] = i1l;
  s2v[wv][lane] = v2l;
  s2i[wv][lane] = i2l;   // same-wave LDS RAW: hw dscnt waits suffice

  unsigned taken = 0u;
  float bs = 0.0f; int ba = 0, bb = 0;
  for (int iter = 0; iter < 32; ++iter) {
    float m = -__builtin_inff(); int mb = -1;
#pragma unroll
    for (int b = 0; b < 32; ++b) {
      if (!((taken >> b) & 1u)) {
        float cc = v1l + s2v[wv][b];
        if (cc > m) { m = cc; mb = b; }
      }
    }
    if (mb < 0) mb = 0;
    int mi = (lane << 5) | mb;
#pragma unroll
    for (int off = 16; off; off >>= 1) {
      float ov = __shfl_xor(m, off, 32);
      int   oi = __shfl_xor(mi, off, 32);
      if (ov > m || (ov == m && oi < mi)) { m = ov; mi = oi; }
    }
    if (lane == (mi >> 5)) taken |= 1u << (mi & 31);
    if (lane == iter) { bs = m; ba = mi >> 5; bb = mi & 31; }
  }

  float mx = __shfl(bs, 0, 32);     // lane 0 holds the max (descending extraction)
  float e  = __expf(bs - mx);
  float sum = e;
#pragma unroll
  for (int off = 16; off; off >>= 1) sum += __shfl_xor(sum, off, 32);
  float w = e / sum;

  int idx = s1i[wv][ba] * NKEYSC + s2i[wv][bb];
  Wgt[(size_t)g * KNNC + lane] = w;
  Idx[(size_t)g * KNNC + lane] = idx;
}

// ---------------- gather + residual + LayerNorm (one block per token) ----------------
__launch_bounds__(256)
__global__ void pkm_output_kernel(const float* __restrict__ DENSE,  // (T,768)
                                  const float* __restrict__ input,  // (T,768)
                                  const float* __restrict__ values, // (16384,768)
                                  const float* __restrict__ Wgt,
                                  const int*   __restrict__ Idx,
                                  const float* __restrict__ gamma,
                                  const float* __restrict__ beta,
                                  float* __restrict__ out) {
  __shared__ float wts[HEADSC * KNNC];
  __shared__ int   ids[HEADSC * KNNC];
  __shared__ float red[256];
  const int t = blockIdx.x, tid = threadIdx.x;

  if (tid < HEADSC * KNNC) {
    wts[tid] = Wgt[(size_t)t * (HEADSC * KNNC) + tid];
    ids[tid] = Idx[(size_t)t * (HEADSC * KNNC) + tid];
  }
  __syncthreads();

  float acc[3];
#pragma unroll
  for (int e = 0; e < 3; ++e) {
    int col = tid + e * 256;
    acc[e] = DENSE[(size_t)t * V_DIMC + col] + input[(size_t)t * V_DIMC + col];
  }
  for (int j = 0; j < HEADSC * KNNC; ++j) {
    if (j + 8 < HEADSC * KNNC)   // global_prefetch of upcoming L2-resident row
      __builtin_prefetch(values + (size_t)ids[j + 8] * V_DIMC + tid, 0, 3);
    float w = wts[j];
    const float* vr = values + (size_t)ids[j] * V_DIMC;
#pragma unroll
    for (int e = 0; e < 3; ++e) acc[e] += w * vr[tid + e * 256];
  }

  red[tid] = acc[0] + acc[1] + acc[2];
  __syncthreads();
  for (int st = 128; st > 0; st >>= 1) {
    if (tid < st) red[tid] += red[tid + st];
    __syncthreads();
  }
  float mu = red[0] * (1.0f / V_DIMC);
  __syncthreads();
  float vs = 0.0f;
#pragma unroll
  for (int e = 0; e < 3; ++e) { float d = acc[e] - mu; vs += d * d; }
  red[tid] = vs;
  __syncthreads();
  for (int st = 128; st > 0; st >>= 1) {
    if (tid < st) red[tid] += red[tid + st];
    __syncthreads();
  }
  float rstd = rsqrtf(red[0] * (1.0f / V_DIMC) + 1e-12f);

#pragma unroll
  for (int e = 0; e < 3; ++e) {
    int col = tid + e * 256;
    out[(size_t)t * V_DIMC + col] = (acc[e] - mu) * rstd * gamma[col] + beta[col];
  }
}

// ---------------- launch ----------------
extern "C" void kernel_launch(void* const* d_in, const int* in_sizes, int n_in,
                              void* d_out, int out_size, void* d_ws, size_t ws_size,
                              hipStream_t stream) {
  const float* hidden = (const float*)d_in[0];
  const float* input  = (const float*)d_in[1];
  const float* Wd     = (const float*)d_in[2];
  const float* bd     = (const float*)d_in[3];
  const float* Wq     = (const float*)d_in[4];
  const float* bq     = (const float*)d_in[5];
  const float* keys   = (const float*)d_in[6];
  const float* values = (const float*)d_in[7];
  const float* gamma  = (const float*)d_in[8];
  const float* beta   = (const float*)d_in[9];
  float* out = (float*)d_out;

  char* ws = (char*)d_ws;
  size_t off = 0;
  auto alloc = [&](size_t bytes) -> void* {
    void* p = ws + off;
    off = (off + bytes + 255) & ~(size_t)255;
    return p;
  };

  __bf16* XB    = (__bf16*)alloc((size_t)T_TOK * IN_DIMC * sizeof(__bf16));
  __bf16* WT    = (__bf16*)alloc((size_t)NCATC * IN_DIMC * sizeof(__bf16));
  __bf16* KB    = (__bf16*)alloc((size_t)HEADSC * 2 * NKEYSC * HALFC * sizeof(__bf16));
  float*  BIAS  = (float*) alloc((size_t)NCATC * sizeof(float));
  float*  DENSE = (float*) alloc((size_t)T_TOK * V_DIMC * sizeof(float));
  __bf16* QB    = (__bf16*)alloc((size_t)T_TOK * QDIMC * sizeof(__bf16));
  float*  SB    = (float*) alloc((size_t)HEADSC * 2 * T_TOK * HALFC * sizeof(float));
  float*  WGT   = (float*) alloc((size_t)T_TOK * HEADSC * KNNC * sizeof(float));
  int*    IDX   = (int*)   alloc((size_t)T_TOK * HEADSC * KNNC * sizeof(int));

  // 1) converts
  {
    int n = T_TOK * IN_DIMC;
    f32_to_bf16_kernel<<<(n + 255) / 256, 256, 0, stream>>>(hidden, XB, n);
  }
  {
    int n = NCATC * IN_DIMC;
    build_wt_kernel<<<(n + 255) / 256, 256, 0, stream>>>(Wd, Wq, WT);
  }
  build_bias_kernel<<<(NCATC + 255) / 256, 256, 0, stream>>>(bd, bq, BIAS);
  {
    int n = HEADSC * 2 * NKEYSC * HALFC;
    f32_to_bf16_kernel<<<(n + 255) / 256, 256, 0, stream>>>(keys, KB, n);
  }

  // 2) fused dense+q GEMM (dense -> fp32 DENSE, q -> bf16 QB, bias fused)
  {
    dim3 grid(T_TOK / BM, NCATC / BN);
    gemm_main_kernel<<<grid, 256, 0, stream>>>(XB, WT, BIAS, DENSE, QB);
  }

  // 3) batched score GEMMs (z = h*2+sub); keys (N,K) already in Bt layout
  {
    dim3 grid(T_TOK / BM, 1, HEADSC * 2);
    gemm_score_kernel<<<grid, 256, 0, stream>>>(QB, KB, SB);
  }

  // 4) top-k + cartesian top-k + softmax (8 waves/block, one wave per (t,h))
  pkm_topk_kernel<<<(T_TOK * HEADSC) / 8, 256, 0, stream>>>(SB, WGT, IDX);

  // 5) gather + residual + LayerNorm
  pkm_output_kernel<<<T_TOK, 256, 0, stream>>>(DENSE, input, values, WGT, IDX,
                                               gamma, beta, out);
}
